// GEN_32830730011096
// MI455X (gfx1250) — matrix operand order repaired and checked by reading, not compiled, not run on hardware
//
#include <hip/hip_runtime.h>
#include <cstdint>

#define BB     2
#define NS_    4096
#define NQ_    4096
#define NN_    8192
#define NE_    131072
#define DH_    128
#define NSTEPS 4

typedef _Float16 v16h __attribute__((ext_vector_type(16)));
typedef _Float16 v8h  __attribute__((ext_vector_type(8)));
typedef _Float16 v4h  __attribute__((ext_vector_type(4)));
typedef float    v8f  __attribute__((ext_vector_type(8)));
typedef float    v4f  __attribute__((ext_vector_type(4)));

constexpr int MODE_ENC = 0, MODE_EDGE = 1, MODE_NODE = 2, MODE_DEC = 3;

// ---- WMMA fragment loaders (CDNA5 ISA 7.12.2 layouts, wave32) ----
// Each fragment = two contiguous 16B LDS loads (ds_load_b128) + shuffle.

__device__ __forceinline__ v16h frag2(const _Float16* p0, const _Float16* p1) {
  v8h lo = *(const v8h*)p0;
  v8h hi = *(const v8h*)p1;
  return __builtin_shufflevector(lo, hi, 0, 1, 2, 3, 4, 5, 6, 7,
                                 8, 9, 10, 11, 12, 13, 14, 15);
}

// A: 16x32 (MxK) f16 tile, LDS row-major with row stride `stride`, K offset k0.
__device__ __forceinline__ v16h load_a_frag(const _Float16* lA, int stride, int k0, int lane) {
  int g = lane >> 4, r = lane & 15;
  const _Float16* base = lA + r * stride + k0 + g * 8;
  return frag2(base, base + 16);
}

// B: 32x16 (KxN) sub-tile of a K-transposed LDS buffer lBt[n*32 + K].
__device__ __forceinline__ v16h load_b_frag(const _Float16* lBt, int n0, int lane) {
  int g = lane >> 4, c = lane & 15;
  const _Float16* base = lBt + (n0 + c) * 32 + g * 8;
  return frag2(base, base + 16);
}

__device__ __forceinline__ v8f wmma16(v16h a, v16h b, v8f c) {
  return __builtin_amdgcn_wmma_f32_16x16x32_f16(false, a, false, b, (short)0, c, false, false);
}

// stage a float4 from global into K-transposed LDS as f16
__device__ __forceinline__ void stage_bt4(_Float16* lBt, int n4, int kk, v4f w) {
  lBt[(n4 + 0) * 32 + kk] = (_Float16)w[0];
  lBt[(n4 + 1) * 32 + kk] = (_Float16)w[1];
  lBt[(n4 + 2) * 32 + kk] = (_Float16)w[2];
  lBt[(n4 + 3) * 32 + kk] = (_Float16)w[3];
}

// ---- generic fused 2-layer MLP: out = relu(A@W1+b1)@W2 + b2 (+residual) ----
// 32-row tiles, 8 waves; each wave owns a 16-col strip and issues 2 WMMAs/chunk.

template <int MODE, int K1, int N2>
__global__ __launch_bounds__(256) void mlp2_kernel(
    int rowsPerBatch,
    const float* __restrict__ in0, const float* __restrict__ in1,
    const int* __restrict__ idxR, const int* __restrict__ idxS,
    const float* __restrict__ W1, const float* __restrict__ b1,
    const float* __restrict__ W2, const float* __restrict__ b2,
    float* __restrict__ outp) {
  __shared__ __align__(16) _Float16 lA[32 * 32];     // A chunk: 32 rows x 32 K
  __shared__ __align__(16) _Float16 lBt[128 * 32];   // weights, K-transposed
  __shared__ __align__(16) _Float16 lH[32 * 128];    // hidden (f16)
  __shared__ int sIdxR[32], sIdxS[32];

  const int tid = threadIdx.x;
  const int lane = tid & 31;
  const int wave = tid >> 5;
  const int bpb = rowsPerBatch / 32;
  const int b = blockIdx.x / bpb;
  const int r0 = (blockIdx.x % bpb) * 32;

  if (MODE == MODE_EDGE) {  // cache endpoint indices once per block
    if (tid < 32) { sIdxR[tid] = idxR[r0 + tid]; sIdxS[tid] = idxS[r0 + tid]; }
  }

  // ---- layer 1 ----
  v8f acc0 = {}, acc1 = {};
  const int nch1 = (K1 + 31) / 32;
  for (int ch = 0; ch < nch1; ++ch) {
    __syncthreads();  // covers sIdx visibility + prior-iter LDS reads
    // stage A (32 rows x 32 k)
    if (MODE == MODE_ENC || (MODE == MODE_DEC && ch == 4)) {
      // scalar path: short/unaligned rows (s: 9-wide, q: 3-wide tail)
      for (int i = tid; i < 1024; i += 256) {
        int rr = i >> 5, kk = i & 31;
        float v = 0.f;
        if (MODE == MODE_ENC) {
          if (kk < 9) v = in0[(size_t)(b * rowsPerBatch + r0 + rr) * 9 + kk];
        } else {
          if (kk < 3) v = in1[(size_t)(b * NQ_ + r0 + rr) * 3 + kk];
        }
        lA[rr * 32 + kk] = (_Float16)v;
      }
    } else {
      // vector path: one float4 (global_load_b128) + one ds_store_b64 per thread
      int rr = tid >> 3, k4 = (tid & 7) * 4;
      const float* src;
      if (MODE == MODE_EDGE) {
        int seg = ch >> 2;
        if (seg == 0)      src = in0 + (size_t)(b * NE_ + r0 + rr) * 128;
        else if (seg == 1) src = in1 + (size_t)(b * NN_ + sIdxR[rr]) * 128;
        else               src = in1 + (size_t)(b * NN_ + sIdxS[rr]) * 128;
        src += (ch & 3) * 32;
      } else if (MODE == MODE_NODE) {
        src = (ch < 4 ? in0 : in1) + (size_t)(b * NN_ + r0 + rr) * 128 + (ch & 3) * 32;
      } else {  // MODE_DEC, ch < 4
        src = in0 + (size_t)(b * NQ_ + r0 + rr) * 128 + ch * 32;
      }
      v4f v = *(const v4f*)(src + k4);
      v4h h;
      h[0] = (_Float16)v[0]; h[1] = (_Float16)v[1];
      h[2] = (_Float16)v[2]; h[3] = (_Float16)v[3];
      *(v4h*)(lA + rr * 32 + k4) = h;
    }
    // stage W1 chunk, K-transposed (float4 global loads, coalesced in n)
    for (int i = tid; i < 1024; i += 256) {
      int n4 = (i & 31) * 4, kk = i >> 5;
      int k = ch * 32 + kk;
      v4f w = {};
      if (k < K1) w = *(const v4f*)(W1 + (size_t)k * 128 + n4);
      stage_bt4(lBt, n4, kk, w);
    }
    __syncthreads();
    v16h bf = load_b_frag(lBt, wave * 16, lane);
    v16h a0 = load_a_frag(lA, 32, 0, lane);
    v16h a1 = load_a_frag(lA + 16 * 32, 32, 0, lane);
    acc0 = wmma16(a0, bf, acc0);
    acc1 = wmma16(a1, bf, acc1);
  }
  __syncthreads();

  // bias + relu -> hidden LDS
  {
    int g = lane >> 4, c = lane & 15, col = wave * 16 + c;
    float bb = b1[col];
#pragma unroll
    for (int r = 0; r < 8; ++r) {
      float h0 = acc0[r] + bb;
      float h1 = acc1[r] + bb;
      lH[(r + 8 * g) * 128 + col] = (_Float16)(h0 > 0.f ? h0 : 0.f);
      lH[(16 + r + 8 * g) * 128 + col] = (_Float16)(h1 > 0.f ? h1 : 0.f);
    }
  }

  // ---- layer 2 (K = 128) ----
  v8f d0 = {}, d1 = {};
  for (int ch = 0; ch < 4; ++ch) {
    __syncthreads();
    if (N2 == 128) {
      for (int i = tid; i < 1024; i += 256) {
        int n4 = (i & 31) * 4, kk = i >> 5;
        v4f w = *(const v4f*)(W2 + (size_t)(ch * 32 + kk) * 128 + n4);
        stage_bt4(lBt, n4, kk, w);
      }
    } else {  // narrow output (decoder head, N2=6): scalar with padding
      for (int i = tid; i < 4096; i += 256) {
        int n = i & 127, kk = i >> 7;
        lBt[n * 32 + kk] =
            (n < N2) ? (_Float16)W2[(size_t)(ch * 32 + kk) * N2 + n] : (_Float16)0.f;
      }
    }
    __syncthreads();
    v16h bf = load_b_frag(lBt, wave * 16, lane);
    v16h a0 = load_a_frag(lH, 128, ch * 32, lane);
    v16h a1 = load_a_frag(lH + 16 * 128, 128, ch * 32, lane);
    d0 = wmma16(a0, bf, d0);
    d1 = wmma16(a1, bf, d1);
  }

  // epilogue: bias (+residual) -> global
  {
    int g = lane >> 4, c = lane & 15, col = wave * 16 + c;
    if (col < N2) {
      float bb = b2[col];
#pragma unroll
      for (int r = 0; r < 8; ++r) {
        int row0 = r0 + r + 8 * g;
        int row1 = row0 + 16;
        float v0 = d0[r] + bb;
        float v1 = d1[r] + bb;
        if (MODE == MODE_NODE) {
          v0 += in0[(size_t)(b * rowsPerBatch + row0) * 128 + col];
          v1 += in0[(size_t)(b * rowsPerBatch + row1) * 128 + col];
        }
        outp[(size_t)(b * rowsPerBatch + row0) * N2 + col] = v0;
        outp[(size_t)(b * rowsPerBatch + row1) * N2 + col] = v1;
      }
    }
  }
}

// ---- online softmax stats over NN_ nodes per point: M = max l, S = sum exp(l-M) ----

__global__ __launch_bounds__(128) void softmax_stats_kernel(
    const float* __restrict__ pts, const float* __restrict__ nodep,
    float* __restrict__ Mout, float* __restrict__ Sout) {
  const int pid = blockIdx.x;
  const float* p = pts + (size_t)pid * 3;
  float px = p[0], py = p[1], pz = p[2];
  float pn2 = px * px + py * py + pz * pz;
  float m = -3.0e38f, s = 0.f;
  for (int n = threadIdx.x; n < NN_; n += 128) {
    float nx = nodep[n * 3 + 0], ny = nodep[n * 3 + 1], nz = nodep[n * 3 + 2];
    float l = 2.f * (px * nx + py * ny + pz * nz) - pn2 - (nx * nx + ny * ny + nz * nz);
    if (l > m) { s = s * __expf(m - l) + 1.f; m = l; }
    else       { s += __expf(l - m); }
  }
  __shared__ float sm[128], ss[128];
  sm[threadIdx.x] = m; ss[threadIdx.x] = s;
  __syncthreads();
  for (int off = 64; off > 0; off >>= 1) {
    if (threadIdx.x < off) {
      float m1 = sm[threadIdx.x], s1 = ss[threadIdx.x];
      float m2 = sm[threadIdx.x + off], s2 = ss[threadIdx.x + off];
      float mm = m1 > m2 ? m1 : m2;
      ss[threadIdx.x] = s1 * __expf(m1 - mm) + s2 * __expf(m2 - mm);
      sm[threadIdx.x] = mm;
    }
    __syncthreads();
  }
  if (threadIdx.x == 0) { Mout[pid] = sm[0]; Sout[pid] = ss[0]; }
}

// ---- fused soft-assignment GEMM: out[b,r,h] = sum_c p(r,c) * V[b,c,h] ----
// p(r,c) = exp(2 r.c - |r|^2 - |c|^2 - M)/S, stats attached to rows or cols.

__global__ __launch_bounds__(256) void aggregate_kernel(
    const float* __restrict__ rowvec, int rv_bstride, int NRpb,
    const float* __restrict__ colvec, int cv_bstride, int NCpb,
    const float* __restrict__ Mst, const float* __restrict__ Sst, int stats_on_col,
    const float* __restrict__ V, float* __restrict__ outp) {
  __shared__ __align__(16) _Float16 lA[32 * 32];
  __shared__ __align__(16) _Float16 lBt[128 * 32];
  __shared__ float rX[32], rY[32], rZ[32], rN2[32], rM[32], rS[32];
  __shared__ float cX[32], cY[32], cZ[32], cN2[32], cM[32], cS[32];

  const int tid = threadIdx.x;
  const int lane = tid & 31;
  const int wave = tid >> 5;
  const int bpb = NRpb / 32;
  const int b = blockIdx.x / bpb;
  const int r0 = (blockIdx.x % bpb) * 32;

  if (tid < 32) {
    const float* rv = rowvec + (size_t)b * rv_bstride + (size_t)(r0 + tid) * 3;
    float x = rv[0], y = rv[1], z = rv[2];
    rX[tid] = x; rY[tid] = y; rZ[tid] = z; rN2[tid] = x * x + y * y + z * z;
    if (!stats_on_col) {
      rM[tid] = Mst[(size_t)b * NRpb + r0 + tid];
      rS[tid] = Sst[(size_t)b * NRpb + r0 + tid];
    }
  }

  v8f acc0 = {}, acc1 = {};
  for (int c0 = 0; c0 < NCpb; c0 += 32) {
    __syncthreads();  // first-iter rX visibility + prior-iter LDS reads
    if (tid < 32) {
      if (c0 + 32 < NCpb)  // prefetch next V chunk (global_prefetch_b8)
        __builtin_prefetch(V + (size_t)(b * NCpb + c0 + 32 + tid) * 128, 0, 1);
      const float* cv = colvec + (size_t)b * cv_bstride + (size_t)(c0 + tid) * 3;
      float x = cv[0], y = cv[1], z = cv[2];
      cX[tid] = x; cY[tid] = y; cZ[tid] = z; cN2[tid] = x * x + y * y + z * z;
      if (stats_on_col) {
        cM[tid] = Mst[(size_t)b * NCpb + c0 + tid];
        cS[tid] = Sst[(size_t)b * NCpb + c0 + tid];
      }
    }
    for (int i = tid; i < 1024; i += 256) {   // stage V chunk (float4) -> f16 transposed
      int n4 = (i & 31) * 4, kk = i >> 5;
      v4f v = *(const v4f*)(V + (size_t)(b * NCpb + c0 + kk) * 128 + n4);
      stage_bt4(lBt, n4, kk, v);
    }
    __syncthreads();
    for (int i = tid; i < 1024; i += 256) {   // on-the-fly score tile 32x32
      int rr = i >> 5, cc = i & 31;
      float l = 2.f * (rX[rr] * cX[cc] + rY[rr] * cY[cc] + rZ[rr] * cZ[cc]) - rN2[rr] - cN2[cc];
      float p = stats_on_col ? (__expf(l - cM[cc]) / cS[cc])
                             : (__expf(l - rM[rr]) / rS[rr]);
      lA[rr * 32 + cc] = (_Float16)p;
    }
    __syncthreads();
    v16h bf = load_b_frag(lBt, wave * 16, lane);
    acc0 = wmma16(load_a_frag(lA, 32, 0, lane), bf, acc0);
    acc1 = wmma16(load_a_frag(lA + 16 * 32, 32, 0, lane), bf, acc1);
  }

  int g = lane >> 4, c = lane & 15, col = wave * 16 + c;
#pragma unroll
  for (int r = 0; r < 8; ++r) {
    outp[(size_t)(b * NRpb + r0 + r + 8 * g) * 128 + col] = acc0[r];
    outp[(size_t)(b * NRpb + r0 + 16 + r + 8 * g) * 128 + col] = acc1[r];
  }
}

// ---- scatter-add messages into receiver nodes ----

__global__ __launch_bounds__(256) void scatter_add_kernel(
    const float* __restrict__ edges, const int* __restrict__ recv,
    float* __restrict__ inbox) {
  size_t gid = (size_t)blockIdx.x * 256 + threadIdx.x;  // b*NE*128 + e*128 + h
  int h = (int)(gid & 127);
  size_t eb = gid >> 7;
  int e = (int)(eb & (NE_ - 1));
  int b = (int)(eb >> 17);
  int n = recv[e];
  atomicAdd(&inbox[(size_t)(b * NN_ + n) * 128 + h], edges[gid]);
}

// ---- host launcher ----

extern "C" void kernel_launch(void* const* d_in, const int* in_sizes, int n_in,
                              void* d_out, int out_size, void* d_ws, size_t ws_size,
                              hipStream_t stream) {
  (void)in_sizes; (void)n_in; (void)out_size; (void)ws_size;
  const float* x         = (const float*)d_in[0];
  const float* s         = (const float*)d_in[1];
  const float* q         = (const float*)d_in[2];
  const int*   senders   = (const int*)d_in[3];
  const int*   receivers = (const int*)d_in[4];
  const float* edges0    = (const float*)d_in[5];
  const float* node_pos  = (const float*)d_in[6];
  const float* enc_W1 = (const float*)d_in[7];
  const float* enc_b1 = (const float*)d_in[8];
  const float* enc_W2 = (const float*)d_in[9];
  const float* enc_b2 = (const float*)d_in[10];
  const float* beW1 = (const float*)d_in[11];
  const float* beb1 = (const float*)d_in[12];
  const float* beW2 = (const float*)d_in[13];
  const float* beb2 = (const float*)d_in[14];
  const float* bnW1 = (const float*)d_in[15];
  const float* bnb1 = (const float*)d_in[16];
  const float* bnW2 = (const float*)d_in[17];
  const float* bnb2 = (const float*)d_in[18];
  const float* dec_W1 = (const float*)d_in[19];
  const float* dec_b1 = (const float*)d_in[20];
  const float* dec_W2 = (const float*)d_in[21];
  const float* dec_b2 = (const float*)d_in[22];
  float* out = (float*)d_out;

  // workspace carve-up (floats): ~160 MB total, edges buffer dominates
  float* ws = (float*)d_ws;
  size_t o = 0;
  float* emb     = ws + o; o += (size_t)BB * NS_ * DH_;
  float* latents = ws + o; o += (size_t)BB * NN_ * DH_;
  float* inbox   = ws + o; o += (size_t)BB * NN_ * DH_;
  float* zbuf    = ws + o; o += (size_t)BB * NQ_ * DH_;
  float* Mx = ws + o; o += (size_t)BB * NS_;
  float* Sx = ws + o; o += (size_t)BB * NS_;
  float* Mq = ws + o; o += (size_t)BB * NQ_;
  float* Sq = ws + o; o += (size_t)BB * NQ_;
  float* edg = ws + o; o += (size_t)BB * NE_ * DH_;

  // 1) encoder MLP: emb = relu(s@W1+b1)@W2+b2   [K=9 -> 128 -> 128]
  mlp2_kernel<MODE_ENC, 9, 128><<<BB * NS_ / 32, 256, 0, stream>>>(
      NS_, s, nullptr, nullptr, nullptr, enc_W1, enc_b1, enc_W2, enc_b2, emb);

  // 2) softmax stats for x and q over NN nodes
  softmax_stats_kernel<<<BB * NS_, 128, 0, stream>>>(x, node_pos, Mx, Sx);
  softmax_stats_kernel<<<BB * NQ_, 128, 0, stream>>>(q, node_pos, Mq, Sq);

  // 3) encoder aggregation: latents[b,n,h] = sum_m sc_x[b,m,n]*emb[b,m,h]
  aggregate_kernel<<<BB * (NN_ / 32), 256, 0, stream>>>(
      node_pos, 0, NN_, x, NS_ * 3, NS_, Mx, Sx, /*stats_on_col=*/1, emb, latents);

  // 4) message-passing steps
  for (int st = 0; st < NSTEPS; ++st) {
    const float* eW1 = beW1 + (size_t)st * 384 * 128;
    const float* eb1 = beb1 + (size_t)st * 128;
    const float* eW2 = beW2 + (size_t)st * 128 * 128;
    const float* eb2 = beb2 + (size_t)st * 128;
    const float* nW1 = bnW1 + (size_t)st * 256 * 128;
    const float* nb1 = bnb1 + (size_t)st * 128;
    const float* nW2 = bnW2 + (size_t)st * 128 * 128;
    const float* nb2 = bnb2 + (size_t)st * 128;

    const float* edges_in = (st == 0) ? edges0 : edg;
    mlp2_kernel<MODE_EDGE, 384, 128><<<BB * NE_ / 32, 256, 0, stream>>>(
        NE_, edges_in, latents, receivers, senders, eW1, eb1, eW2, eb2, edg);

    hipMemsetAsync(inbox, 0, (size_t)BB * NN_ * DH_ * sizeof(float), stream);
    scatter_add_kernel<<<(int)((BB * (size_t)NE_ * DH_) / 256), 256, 0, stream>>>(
        edg, receivers, inbox);

    mlp2_kernel<MODE_NODE, 256, 128><<<BB * NN_ / 32, 256, 0, stream>>>(
        NN_, latents, inbox, nullptr, nullptr, nW1, nb1, nW2, nb2, latents);
  }

  // 5) decoder aggregation: z[b,q,h] = sum_n sc_q[b,q,n]*latents[b,n,h]
  aggregate_kernel<<<BB * (NQ_ / 32), 256, 0, stream>>>(
      q, NQ_ * 3, NQ_, node_pos, 0, NN_, Mq, Sq, /*stats_on_col=*/0, latents, zbuf);

  // 6) decoder MLP: out = relu([z,q]@W1+b1)@W2+b2   [K=131 -> 128 -> 6]
  mlp2_kernel<MODE_DEC, 131, 6><<<BB * NQ_ / 32, 256, 0, stream>>>(
      NQ_, zbuf, q, nullptr, nullptr, dec_W1, dec_b1, dec_W2, dec_b2, out);
}
